// BitNetTransformerMLP_11888469475631
// MI455X (gfx1250) — 8-bit matrix operand order repaired and checked by reading, compiled, not run on hardware
//
#include <hip/hip_runtime.h>
#include <hip/hip_bf16.h>
#include <math.h>
#include <stdint.h>

typedef __attribute__((ext_vector_type(8))) int v8i;
typedef int v4i_b __attribute__((vector_size(16)));

#define EPSQ 1e-5f

#if defined(__has_builtin)
#if __has_builtin(__builtin_amdgcn_global_load_async_to_lds_b128)
#define USE_ASYNC_LDS 1
#endif
#endif

__global__ void absum_part_kernel(const float* __restrict__ w, size_t n4,
                                  float* __restrict__ part) {
  __shared__ float red[256];
  const size_t stride = (size_t)gridDim.x * blockDim.x;
  const float4* w4 = (const float4*)w;
  float s = 0.f;
  for (size_t i = (size_t)blockIdx.x * blockDim.x + threadIdx.x; i < n4; i += stride) {
    float4 v = w4[i];
    s += fabsf(v.x) + fabsf(v.y) + fabsf(v.z) + fabsf(v.w);
  }
  red[threadIdx.x] = s;
  __syncthreads();
  for (int off = 128; off > 0; off >>= 1) {
    if (threadIdx.x < off) red[threadIdx.x] += red[threadIdx.x + off];
    __syncthreads();
  }
  if (threadIdx.x == 0) part[blockIdx.x] = red[0];
}

__global__ void absum_fin_kernel(const float* __restrict__ part, int nparts,
                                 float inv_count, float* __restrict__ swp) {
  __shared__ float red[256];
  float s = 0.f;
  for (int i = threadIdx.x; i < nparts; i += 256) s += part[i];
  red[threadIdx.x] = s;
  __syncthreads();
  for (int off = 128; off > 0; off >>= 1) {
    if (threadIdx.x < off) red[threadIdx.x] += red[threadIdx.x + off];
    __syncthreads();
  }
  if (threadIdx.x == 0) swp[0] = fmaxf(red[0] * inv_count, EPSQ);
}

__global__ void quantw_kernel(const float* __restrict__ w, size_t n4,
                              const float* __restrict__ swp,
                              signed char* __restrict__ wq) {
  const float inv = 1.f / swp[0];
  const size_t stride = (size_t)gridDim.x * blockDim.x;
  const float4* w4 = (const float4*)w;
  char4* q4 = (char4*)wq;
  for (size_t i = (size_t)blockIdx.x * blockDim.x + threadIdx.x; i < n4; i += stride) {
    float4 v = w4[i];
    char4 o;
    o.x = (signed char)(int)fminf(fmaxf(rintf(v.x * inv), -1.f), 1.f);
    o.y = (signed char)(int)fminf(fmaxf(rintf(v.y * inv), -1.f), 1.f);
    o.z = (signed char)(int)fminf(fmaxf(rintf(v.z * inv), -1.f), 1.f);
    o.w = (signed char)(int)fminf(fmaxf(rintf(v.w * inv), -1.f), 1.f);
    q4[i] = o;
  }
}

__global__ void quanta_kernel(const float* __restrict__ x, int K,
                              signed char* __restrict__ xq,
                              float* __restrict__ sx) {
  __shared__ float red[256];
  const int row = blockIdx.x;
  const int tid = threadIdx.x;
  const float4* xr = (const float4*)(x + (size_t)row * K);
  const int n4 = K >> 2;
  float m = 0.f;
  for (int i = tid; i < n4; i += 256) {
    float4 v = xr[i];
    m = fmaxf(m, fmaxf(fmaxf(fabsf(v.x), fabsf(v.y)), fmaxf(fabsf(v.z), fabsf(v.w))));
  }
  red[tid] = m;
  __syncthreads();
  for (int off = 128; off > 0; off >>= 1) {
    if (tid < off) red[tid] = fmaxf(red[tid], red[tid + off]);
    __syncthreads();
  }
  const float s = fmaxf(red[0], EPSQ) * (1.f / 127.f);
  if (tid == 0) sx[row] = s;
  const float inv = 1.f / s;
  char4* q4 = (char4*)(xq + (size_t)row * K);
  for (int i = tid; i < n4; i += 256) {
    float4 v = xr[i];
    char4 o;
    o.x = (signed char)(int)fminf(fmaxf(rintf(v.x * inv), -128.f), 127.f);
    o.y = (signed char)(int)fminf(fmaxf(rintf(v.y * inv), -128.f), 127.f);
    o.z = (signed char)(int)fminf(fmaxf(rintf(v.z * inv), -128.f), 127.f);
    o.w = (signed char)(int)fminf(fmaxf(rintf(v.w * inv), -128.f), 127.f);
    q4[i] = o;
  }
}

__device__ __forceinline__ void async_copy16(const void* g, void* l) {
#if USE_ASYNC_LDS
  __builtin_amdgcn_global_load_async_to_lds_b128(
      (__attribute__((address_space(1))) v4i_b*)(uintptr_t)g,
      (__attribute__((address_space(3))) v4i_b*)(uint32_t)(uintptr_t)l, 0, 0);
#else
  (void)g; (void)l;
#endif
}

__device__ __forceinline__ void wait_async0() {
#if defined(__has_builtin)
#if __has_builtin(__builtin_amdgcn_s_wait_asynccnt)
  __builtin_amdgcn_s_wait_asynccnt(0);
  return;
#endif
#endif
  asm volatile("s_wait_asynccnt 0x0" ::: "memory");
}

template <bool GELU>
__global__ __launch_bounds__(256) void bitgemm_kernel(
    const signed char* __restrict__ A, const signed char* __restrict__ Bw,
    const float* __restrict__ sx, const float* __restrict__ swp,
    float* __restrict__ C, int M, int N, int K) {
  __shared__ uint4 smA[2][512];
  __shared__ uint4 smB[2][512];

  const int tid = threadIdx.x;
  const int lane = tid & 31;
  const int l16 = lane & 15;
  const int half = lane >> 4;
  const int wv = tid >> 5;
  const int wm = wv & 1;
  const int wn = wv >> 1;
  const int bm = blockIdx.y, bn = blockIdx.x;
  const int nk = K >> 6;

  const size_t rowA0 = (size_t)(bm * 128) * (size_t)K;
  const size_t rowB0 = (size_t)(bn * 128) * (size_t)K;

  const int lr = tid >> 2;
  const int lc = (tid & 3) << 4;

  const v8i vzero = {0, 0, 0, 0, 0, 0, 0, 0};
  v8i acc[4][2];
#pragma unroll
  for (int mt = 0; mt < 4; ++mt)
#pragma unroll
    for (int nt = 0; nt < 2; ++nt) acc[mt][nt] = vzero;

#if USE_ASYNC_LDS
  auto issue = [&](int k0, int buf) {
#pragma unroll
    for (int i = 0; i < 2; ++i) {
      const int r = lr + i * 64;
      async_copy16(A + rowA0 + (size_t)r * K + k0 + lc,
                   (char*)smA[buf] + r * 64 + lc);
      async_copy16(Bw + rowB0 + (size_t)r * K + k0 + lc,
                   (char*)smB[buf] + r * 64 + lc);
    }
  };
#else
  uint4 ra[2], rb[2];
  auto gload = [&](int k0) {
#pragma unroll
    for (int i = 0; i < 2; ++i) {
      const int r = lr + i * 64;
      ra[i] = *(const uint4*)(A + rowA0 + (size_t)r * K + k0 + lc);
      rb[i] = *(const uint4*)(Bw + rowB0 + (size_t)r * K + k0 + lc);
    }
  };
  auto sstore = [&](int buf) {
#pragma unroll
    for (int i = 0; i < 2; ++i) {
      const int r = lr + i * 64;
      *(uint4*)((char*)smA[buf] + r * 64 + lc) = ra[i];
      *(uint4*)((char*)smB[buf] + r * 64 + lc) = rb[i];
    }
  };
#endif

  auto compute = [&](int buf) {
    const char* Ab = (const char*)smA[buf];
    const char* Bb = (const char*)smB[buf];
    v8i afr[4];
#pragma unroll
    for (int mt = 0; mt < 4; ++mt) {
      const char* base = Ab + (wm * 64 + mt * 16 + l16) * 64 + half * 16;
#pragma unroll
      for (int g = 0; g < 2; ++g) {
        uint4 d = *(const uint4*)(base + g * 32);
        afr[mt][4 * g] = (int)d.x;
        afr[mt][4 * g + 1] = (int)d.y;
        afr[mt][4 * g + 2] = (int)d.z;
        afr[mt][4 * g + 3] = (int)d.w;
      }
    }
    v8i bfr[2];
#pragma unroll
    for (int nt = 0; nt < 2; ++nt) {
      const char* base = Bb + (wn * 32 + nt * 16 + l16) * 64 + half * 16;
#pragma unroll
      for (int h = 0; h < 2; ++h) {
        uint4 d = *(const uint4*)(base + h * 32);
        bfr[nt][4 * h] = (int)d.x;
        bfr[nt][4 * h + 1] = (int)d.y;
        bfr[nt][4 * h + 2] = (int)d.z;
        bfr[nt][4 * h + 3] = (int)d.w;
      }
    }
#pragma unroll
    for (int mt = 0; mt < 4; ++mt)
#pragma unroll
      for (int nt = 0; nt < 2; ++nt)
        acc[mt][nt] = __builtin_amdgcn_wmma_i32_16x16x64_iu8(
             true, afr[mt],  true, bfr[nt], acc[mt][nt],
             false,  false);
  };

#if USE_ASYNC_LDS
  issue(0, 0);
  int buf = 0;
  for (int kk = 0; kk < nk; ++kk) {
    wait_async0();
    __syncthreads();
    if (kk + 1 < nk) issue((kk + 1) << 6, buf ^ 1);
    compute(buf);
    buf ^= 1;
  }
#else
  gload(0);
  sstore(0);
  __syncthreads();
  int buf = 0;
  for (int kk = 0; kk < nk; ++kk) {
    if (kk + 1 < nk) gload((kk + 1) << 6);
    compute(buf);
    if (kk + 1 < nk) {
      sstore(buf ^ 1);
      __syncthreads();
      buf ^= 1;
    }
  }
#endif

  const float sw = swp[0];
#pragma unroll
  for (int mt = 0; mt < 4; ++mt) {
#pragma unroll
    for (int nt = 0; nt < 2; ++nt) {
      const int col = bn * 128 + wn * 32 + nt * 16 + l16;
#pragma unroll
      for (int r = 0; r < 8; ++r) {
        const int row = bm * 128 + wm * 64 + mt * 16 + r + half * 8;
        float v = (float)acc[mt][nt][r] * sx[row] * sw;
        if (GELU) v = 0.5f * v * (1.0f + erff(v * 0.70710678118654752f));
        C[(size_t)row * N + col] = v;
      }
    }
  }
}

extern "C" void kernel_launch(void* const* d_in, const int* in_sizes, int n_in,
                              void* d_out, int out_size, void* d_ws, size_t ws_size,
                              hipStream_t stream) {
  const float* x = (const float*)d_in[0];
  const float* W1 = (const float*)d_in[1];
  const float* W2 = (const float*)d_in[2];
  float* out = (float*)d_out;

  const int H = 4096, I = 16384;
  const int M = in_sizes[0] / H;

  char* ws = (char*)d_ws;
  size_t off = 0;
  auto walloc = [&](size_t bytes) -> char* {
    char* p = ws + off;
    off = (off + bytes + 255) & ~(size_t)255;
    return p;
  };
  signed char* wq1 = (signed char*)walloc((size_t)I * H);
  signed char* wq2 = (signed char*)walloc((size_t)H * I);
  signed char* xq1 = (signed char*)walloc((size_t)M * H);
  signed char* hq  = (signed char*)walloc((size_t)M * I);
  float* h   = (float*)walloc((size_t)M * I * sizeof(float));
  float* sx1 = (float*)walloc((size_t)M * sizeof(float));
  float* sx2 = (float*)walloc((size_t)M * sizeof(float));
  float* part = (float*)walloc(4096 * sizeof(float));
  float* sw1 = (float*)walloc(sizeof(float));
  float* sw2 = (float*)walloc(sizeof(float));
  (void)ws_size;
  (void)n_in;
  (void)out_size;

  const size_t nW = (size_t)I * H;

  absum_part_kernel<<<4096, 256, 0, stream>>>(W1, nW / 4, part);
  absum_fin_kernel<<<1, 256, 0, stream>>>(part, 4096, 1.f / (float)nW, sw1);
  quantw_kernel<<<4096, 256, 0, stream>>>(W1, nW / 4, sw1, wq1);

  absum_part_kernel<<<4096, 256, 0, stream>>>(W2, nW / 4, part);
  absum_fin_kernel<<<1, 256, 0, stream>>>(part, 4096, 1.f / (float)nW, sw2);
  quantw_kernel<<<4096, 256, 0, stream>>>(W2, nW / 4, sw2, wq2);

  quanta_kernel<<<M, 256, 0, stream>>>(x, H, xq1, sx1);
  bitgemm_kernel<true><<<dim3(I / 128, M / 128), 256, 0, stream>>>(
      xq1, wq1, sx1, sw1, h, M, I, H);

  quanta_kernel<<<M, 256, 0, stream>>>(h, I, hq, sx2);
  bitgemm_kernel<false><<<dim3(H / 128, M / 128), 256, 0, stream>>>(
      hq, wq2, sx2, sw2, out, M, H, I);
}
